// LSHAttentionLayer_74869869904192
// MI455X (gfx1250) — matrix operand order repaired and checked by reading, not compiled, not run on hardware
//
#include <hip/hip_runtime.h>
#include <hip/hip_bf16.h>

#define NROWS 8192
#define DFEAT 64
#define KFEAT 512
#define NEGINF -9e15f

typedef __bf16 bf16_t;
typedef __attribute__((ext_vector_type(8)))  bf16_t bf16x8;
typedef __attribute__((ext_vector_type(16))) bf16_t bf16x16;
typedef __attribute__((ext_vector_type(4)))  float  f32x4;
typedef __attribute__((ext_vector_type(8)))  float  f32x8;

// ---------------------------------------------------------------- helpers ---
__device__ __forceinline__ f32x8 wmma_bf16(bf16x16 a, bf16x16 b, f32x8 c) {
  // D = A(16x32 bf16) * B(32x16 bf16) + C(16x16 f32)
  return __builtin_amdgcn_wmma_f32_16x16x32_bf16(
      false, a, false, b, (short)0, c, false, false);
}

__device__ __forceinline__ bf16x16 cat16(bf16x8 lo, bf16x8 hi) {
  return __builtin_shufflevector(lo, hi, 0,1,2,3,4,5,6,7,8,9,10,11,12,13,14,15);
}

// A-fragment (16x32, 16-bit): lane half h holds K = {h*8..h*8+7, 16+h*8..}
__device__ __forceinline__ bf16x16 frag_a_bf(const bf16_t* rowp, int k0, int kh8) {
  bf16x8 lo = *(const bf16x8*)(rowp + k0 + kh8);
  bf16x8 hi = *(const bf16x8*)(rowp + k0 + kh8 + 16);
  return cat16(lo, hi);
}

// A-fragment built from f32 memory with on-the-fly bf16 conversion
__device__ __forceinline__ bf16x16 frag_a_f32(const float* rowp, int k0, int kh8) {
  f32x4 a0 = *(const f32x4*)(rowp + k0 + kh8);
  f32x4 a1 = *(const f32x4*)(rowp + k0 + kh8 + 4);
  f32x4 a2 = *(const f32x4*)(rowp + k0 + kh8 + 16);
  f32x4 a3 = *(const f32x4*)(rowp + k0 + kh8 + 20);
  f32x8 lo = __builtin_shufflevector(a0, a1, 0,1,2,3,4,5,6,7);
  f32x8 hi = __builtin_shufflevector(a2, a3, 0,1,2,3,4,5,6,7);
  return cat16(__builtin_convertvector(lo, bf16x8),
               __builtin_convertvector(hi, bf16x8));
}

// B-fragment (32x16, 16-bit): lane n holds K of column n,
// half h holds K = {h*8..h*8+7, 16+h*8..}.  colp points at column n's K-contiguous data.
__device__ __forceinline__ bf16x16 frag_b_bf(const bf16_t* colp, int k0, int hf) {
  const bf16_t* p = colp + k0 + hf * 8;
  return cat16(*(const bf16x8*)p, *(const bf16x8*)(p + 16));
}

// reduce across the 16 lanes of this lane's half (wave32)
__device__ __forceinline__ float rmax16(float v) {
  v = fmaxf(v, __shfl_xor(v, 1, 32));
  v = fmaxf(v, __shfl_xor(v, 2, 32));
  v = fmaxf(v, __shfl_xor(v, 4, 32));
  v = fmaxf(v, __shfl_xor(v, 8, 32));
  return v;
}
__device__ __forceinline__ float rsum16(float v) {
  v += __shfl_xor(v, 1, 32);
  v += __shfl_xor(v, 2, 32);
  v += __shfl_xor(v, 4, 32);
  v += __shfl_xor(v, 8, 32);
  return v;
}

// ------------------------------------------------------------ kernel 0 ------
// transpose + convert weights: [512,64] f32 -> [64,512] bf16
__global__ __launch_bounds__(256) void prep_weights(
    const float* __restrict__ kW, const float* __restrict__ vW,
    bf16_t* __restrict__ kWT, bf16_t* __restrict__ vWT) {
  int i = blockIdx.x * 256 + threadIdx.x;   // 0 .. 512*64-1
  int k = i >> 6, n = i & 63;
  kWT[n * KFEAT + k] = (bf16_t)kW[i];
  vWT[n * KFEAT + k] = (bf16_t)vW[i];
}

// ------------------------------------------------------------ kernel 1 ------
// kh = X@kW -> bf16 [8192,64]; vh = X@vW -> stored transposed bf16 [64,8192]
__global__ __launch_bounds__(256) void proj_kernel(
    const float* __restrict__ input, const bf16_t* __restrict__ kWT,
    const bf16_t* __restrict__ vWT, bf16_t* __restrict__ khb,
    bf16_t* __restrict__ vhT) {
  __shared__ bf16_t vt_s[64][136];          // padded: conflict-free transpose
  const int tid = threadIdx.x;
  const int w = tid >> 5, lane = tid & 31;
  const int n = lane & 15, hf = lane >> 4;
  const int rowTop = blockIdx.x * 128;      // 8 waves x 16 rows
  const float* arow = input + (size_t)(rowTop + w * 16 + n) * KFEAT;

  f32x8 ck[4], cv[4];
  const f32x8 zero = {0.f,0.f,0.f,0.f,0.f,0.f,0.f,0.f};
#pragma unroll
  for (int t = 0; t < 4; ++t) { ck[t] = zero; cv[t] = zero; }

  for (int k0 = 0; k0 < KFEAT; k0 += 32) {
    bf16x16 a = frag_a_f32(arow, k0, hf * 8);
#pragma unroll
    for (int t = 0; t < 4; ++t) {
      const int col = t * 16 + n;
      ck[t] = wmma_bf16(a, frag_b_bf(kWT + (size_t)col * KFEAT, k0, hf), ck[t]);
      cv[t] = wmma_bf16(a, frag_b_bf(vWT + (size_t)col * KFEAT, k0, hf), cv[t]);
    }
  }

#pragma unroll
  for (int t = 0; t < 4; ++t) {
#pragma unroll
    for (int v = 0; v < 8; ++v) {
      const int rloc = w * 16 + v + 8 * hf;
      khb[(size_t)(rowTop + rloc) * DFEAT + t * 16 + n] = (bf16_t)ck[t][v];
      vt_s[t * 16 + n][rloc] = (bf16_t)cv[t][v];     // transpose via LDS
    }
  }
  __syncthreads();
  {            // coalesced store of vh^T: 64 feature-rows x 128 contiguous cols
    const int cc = tid >> 2;
    const int off = (tid & 3) * 32;
    const bf16x8* s = (const bf16x8*)(&vt_s[cc][off]);
    bf16x8* d = (bf16x8*)(vhT + (size_t)cc * NROWS + rowTop + off);
    d[0] = s[0]; d[1] = s[1]; d[2] = s[2]; d[3] = s[3];
  }
}

// ------------------------------------------------------------ kernel 2 ------
// fused masked attention: online softmax over column blocks of 64, all WMMA.
// K/V blocks staged memory->LDS with GLOBAL_LOAD_ASYNC_TO_LDS_B128 (ASYNCcnt),
// double-buffered so the next block's DMA overlaps this block's WMMAs.
__global__ __launch_bounds__(256) void attn_kernel(
    const bf16_t* __restrict__ khb, const bf16_t* __restrict__ vhT,
    const int* __restrict__ adj, float* __restrict__ out) {
  __shared__ bf16_t kh_s[2][64][72];        // K blocks (double-buffered)
  __shared__ bf16_t v_s[2][64][72];         // V blocks (vh^T rows, K-contig)
  __shared__ bf16_t p_s[8][16][72];         // per-wave P tile round-trip

  const int tid = threadIdx.x;
  const int w = tid >> 5, lane = tid & 31;
  const int n = lane & 15, hf = lane >> 4;
  const int rowTop = blockIdx.x * 128 + w * 16;

  const int ldr = tid >> 2;                 // 0..63
  const int ldc = (tid & 3) << 4;           // 0,16,32,48

  // async DMA of one 64x64 bf16 K block + V block into LDS buffer `buf`
  // (INST_OFFSET is added to BOTH the LDS and global addresses)
  auto stage = [&](int jb, int buf) {
    const unsigned lk =
        (unsigned)(unsigned long long)(const void*)&kh_s[buf][ldr][ldc];
    const unsigned long long gk =
        (unsigned long long)(const void*)(khb + (size_t)(jb + ldr) * DFEAT + ldc);
    asm volatile("global_load_async_to_lds_b128 %0, %1, off"
                 :: "v"(lk), "v"(gk) : "memory");
    asm volatile("global_load_async_to_lds_b128 %0, %1, off offset:16"
                 :: "v"(lk), "v"(gk) : "memory");
    const unsigned lv =
        (unsigned)(unsigned long long)(const void*)&v_s[buf][ldr][ldc];
    const unsigned long long gv =
        (unsigned long long)(const void*)(vhT + (size_t)ldr * NROWS + jb + ldc);
    asm volatile("global_load_async_to_lds_b128 %0, %1, off"
                 :: "v"(lv), "v"(gv) : "memory");
    asm volatile("global_load_async_to_lds_b128 %0, %1, off offset:16"
                 :: "v"(lv), "v"(gv) : "memory");
  };

  // Q fragments for this wave's 16 rows (kh in bf16, K=64 -> 2 fragments)
  const bf16_t* qrow = khb + (size_t)(rowTop + n) * DFEAT;
  const bf16x16 qa0 = frag_a_bf(qrow, 0, hf * 8);
  const bf16x16 qa1 = frag_a_bf(qrow, 32, hf * 8);

  const f32x8 zero = {0.f,0.f,0.f,0.f,0.f,0.f,0.f,0.f};
  f32x8 acc[4];
  float m[8], l[8];
#pragma unroll
  for (int t = 0; t < 4; ++t) acc[t] = zero;
#pragma unroll
  for (int v = 0; v < 8; ++v) { m[v] = NEGINF; l[v] = 0.f; }

  stage(0, 0);                              // prime the pipeline

  for (int jb = 0; jb < NROWS; jb += 64) {
    const int buf = (jb >> 6) & 1;
    // my async loads for `buf` are done; barrier => all waves' loads done and
    // everyone has finished reading buf^1 (so it can be overwritten next)
    asm volatile("s_wait_asynccnt 0x0" ::: "memory");
    __syncthreads();
    if (jb + 64 < NROWS) {
      stage(jb + 64, buf ^ 1);              // DMA next block during compute
      __builtin_prefetch(adj + (size_t)(rowTop + n) * NROWS + jb + 64 + hf * 32,
                         0, 1);             // pull next adj block toward L2
    }

    // ---- S = Q K^T  (B column n == kh row n: no transpose needed) ----
    f32x8 s[4];
#pragma unroll
    for (int t = 0; t < 4; ++t) {
      const bf16_t* kcol = &kh_s[buf][t * 16 + n][0];
      f32x8 c = zero;
      c = wmma_bf16(qa0, frag_b_bf(kcol, 0, hf), c);
      c = wmma_bf16(qa1, frag_b_bf(kcol, 32, hf), c);
      s[t] = c;
    }

    // ---- scale + adj mask + online softmax (C layout: row = v + 8*hf) ----
    float bm[8];
#pragma unroll
    for (int v = 0; v < 8; ++v) bm[v] = NEGINF;
#pragma unroll
    for (int t = 0; t < 4; ++t) {
#pragma unroll
      for (int v = 0; v < 8; ++v) {
        const size_t r = (size_t)(rowTop + v + 8 * hf);
        const int a = adj[r * NROWS + jb + t * 16 + n];
        const float sv = (a > 0) ? s[t][v] * 0.125f : NEGINF;
        s[t][v] = sv;
        bm[v] = fmaxf(bm[v], sv);
      }
    }
    float corr[8], rs[8];
#pragma unroll
    for (int v = 0; v < 8; ++v) {
      const float nm = fmaxf(m[v], rmax16(bm[v]));
      corr[v] = __expf(m[v] - nm);
      m[v] = nm;
      rs[v] = 0.f;
    }
#pragma unroll
    for (int t = 0; t < 4; ++t) {
#pragma unroll
      for (int v = 0; v < 8; ++v) {
        const float p = __expf(s[t][v] - m[v]);
        s[t][v] = p;
        rs[v] += p;
      }
    }
#pragma unroll
    for (int v = 0; v < 8; ++v) l[v] = l[v] * corr[v] + rsum16(rs[v]);
#pragma unroll
    for (int t = 0; t < 4; ++t)
#pragma unroll
      for (int v = 0; v < 8; ++v) acc[t][v] *= corr[v];

    // ---- P (C layout f32) -> LDS -> A-fragment layout bf16 ----
#pragma unroll
    for (int t = 0; t < 4; ++t)
#pragma unroll
      for (int v = 0; v < 8; ++v)
        p_s[w][v + 8 * hf][t * 16 + n] = (bf16_t)s[t][v];
    asm volatile("s_wait_dscnt 0x0" ::: "memory");   // cross-lane LDS RAW

    const bf16_t* prow = &p_s[w][n][0];
    const bf16x16 pa0 = frag_a_bf(prow, 0, hf * 8);
    const bf16x16 pa1 = frag_a_bf(prow, 32, hf * 8);

    // ---- acc += P @ V  (V columns are vh^T rows in v_s: K-contiguous) ----
#pragma unroll
    for (int t = 0; t < 4; ++t) {
      const bf16_t* vcol = &v_s[buf][t * 16 + n][0];
      acc[t] = wmma_bf16(pa0, frag_b_bf(vcol, 0, hf), acc[t]);
      acc[t] = wmma_bf16(pa1, frag_b_bf(vcol, 32, hf), acc[t]);
    }
  }

  // ---- epilogue: normalize + ELU, coalesced f32 stores ----
#pragma unroll
  for (int t = 0; t < 4; ++t) {
#pragma unroll
    for (int v = 0; v < 8; ++v) {
      const float x = acc[t][v] / l[v];
      const float y = (x > 0.f) ? x : expm1f(x);
      out[(size_t)(rowTop + v + 8 * hf) * DFEAT + t * 16 + n] = y;
    }
  }
}

// ------------------------------------------------------------- launcher -----
extern "C" void kernel_launch(void* const* d_in, const int* in_sizes, int n_in,
                              void* d_out, int out_size, void* d_ws, size_t ws_size,
                              hipStream_t stream) {
  (void)in_sizes; (void)n_in; (void)out_size; (void)ws_size;
  const float* input = (const float*)d_in[0];
  const int*   adj   = (const int*)d_in[1];
  const float* kW    = (const float*)d_in[2];
  const float* vW    = (const float*)d_in[3];
  float* out = (float*)d_out;

  char* ws = (char*)d_ws;
  const size_t wbytes = (size_t)DFEAT * KFEAT * sizeof(bf16_t);   // 64KB
  const size_t hbytes = (size_t)NROWS * DFEAT * sizeof(bf16_t);   // 1MB
  bf16_t* kWT = (bf16_t*)ws;
  bf16_t* vWT = (bf16_t*)(ws + wbytes);
  bf16_t* khb = (bf16_t*)(ws + 2 * wbytes);
  bf16_t* vhT = (bf16_t*)(ws + 2 * wbytes + hbytes);

  prep_weights<<<(KFEAT * DFEAT) / 256, 256, 0, stream>>>(kW, vW, kWT, vWT);
  proj_kernel<<<NROWS / 128, 256, 0, stream>>>(input, kWT, vWT, khb, vhT);
  attn_kernel<<<NROWS / 128, 256, 0, stream>>>(khb, vhT, adj, out);
}
